// DGCNNCls_76158360092794
// MI455X (gfx1250) — matrix operand order repaired and checked by reading, not compiled, not run on hardware
//
#include <hip/hip_runtime.h>

typedef __attribute__((ext_vector_type(16))) _Float16 v16h;
typedef __attribute__((ext_vector_type(8)))  float    v8f;

#define NPTS   4096
#define BATCH  4
#define KNN    32
#define FLT_BIG 3.402823466e+38f

#define ACT_NONE  0
#define ACT_RELU  1
#define ACT_LRELU 2

__device__ __forceinline__ float apply_act(float v, int act) {
    if (act == ACT_RELU)  return v > 0.f ? v : 0.f;
    if (act == ACT_LRELU) return v > 0.f ? v : 0.01f * v;
    return v;
}

// ---------------------------------------------------------------------------
// f32 -> f16 conversion (one-time weight pre-conversion)
// ---------------------------------------------------------------------------
__global__ void k_cvt_f16(const float* __restrict__ in, _Float16* __restrict__ out, int n) {
    int t = blockIdx.x * blockDim.x + threadIdx.x;
    if (t < n) out[t] = (_Float16)in[t];
}

// ---------------------------------------------------------------------------
// Generic small linear: out[r,o] = act( (sum_k A[r,k]*W[o,k] + bias[o]) *g[o]+b[o] )
// ---------------------------------------------------------------------------
__global__ void k_linear(const float* __restrict__ A, const float* __restrict__ W,
                         const float* __restrict__ bias, const float* __restrict__ gamma,
                         const float* __restrict__ beta, float* __restrict__ out,
                         int M, int N0, int Kd, int act) {
    int t = blockIdx.x * blockDim.x + threadIdx.x;
    if (t >= M * N0) return;
    int r = t / N0, o = t - r * N0;
    const float* a = A + (size_t)r * Kd;
    const float* w = W + (size_t)o * Kd;
    float s = 0.f;
    for (int k = 0; k < Kd; ++k) s += a[k] * w[k];
    if (bias)  s += bias[o];
    if (gamma) s = s * gamma[o] + beta[o];
    out[(size_t)r * N0 + o] = apply_act(s, act);
}

// ---------------------------------------------------------------------------
// WMMA GEMM: out = act( (A[M,Kd] @ W[N0,Kd]^T) * gamma + beta ), Kd<=128, Kd%32==0
// block = 128 threads (4 waves); tile: M=64 (16 rows/wave), N=16.
// ---------------------------------------------------------------------------
__global__ __launch_bounds__(128) void k_gemm_wmma(
    const float* __restrict__ A, const float* __restrict__ W,
    const float* __restrict__ gamma, const float* __restrict__ beta,
    float* __restrict__ out, int M, int N0, int Kd, int act) {
    __shared__ _Float16 As[64 * 128];
    __shared__ _Float16 Ws[16 * 128];
    const int m0 = blockIdx.x * 64;
    const int n0 = blockIdx.y * 16;
    const int tid = threadIdx.x;

    for (int e = tid; e < 64 * Kd; e += 128) {
        int r = e / Kd, c = e - r * Kd;
        As[r * Kd + c] = (_Float16)A[(size_t)(m0 + r) * Kd + c];
    }
    for (int e = tid; e < 16 * Kd; e += 128) {
        int r = e / Kd, c = e - r * Kd;
        Ws[r * Kd + c] = (_Float16)W[(size_t)(n0 + r) * Kd + c];
    }
    __syncthreads();

    const int wave = tid >> 5, lane = tid & 31;
    const int half = lane >> 4, lr = lane & 15;
    const int mrow = wave * 16 + lr;

    v8f acc = {};
    for (int k0 = 0; k0 < Kd; k0 += 32) {
        v16h a, bf;
#pragma unroll
        for (int j = 0; j < 8; ++j) {
            // A 16-bit 16x32 layout: vgpr0-3 K{0..7 | 8..15}, vgpr4-7 K{16..23 | 24..31}
            int kp = ((j < 4) ? 2 * j : 16 + 2 * (j - 4)) + half * 8;
            a[2 * j]     = As[mrow * Kd + k0 + kp];
            a[2 * j + 1] = As[mrow * Kd + k0 + kp + 1];
            // B 32x16 layout: same K order as A, N=lane%16
            int kb = kp;
            bf[2 * j]     = Ws[lr * Kd + k0 + kb];
            bf[2 * j + 1] = Ws[lr * Kd + k0 + kb + 1];
        }
        acc = __builtin_amdgcn_wmma_f32_16x16x32_f16(false, a, false, bf,
                                                     (short)0, acc, false, false);
    }

    const int gcol = n0 + lr;
    const float g  = gamma ? gamma[gcol] : 1.f;
    const float bb = beta  ? beta[gcol]  : 0.f;
#pragma unroll
    for (int i = 0; i < 8; ++i) {
        int grow = m0 + wave * 16 + half * 8 + i;  // C/D layout: vgpr i -> M = half*8+i
        float v = apply_act(acc[i] * g + bb, act);
        out[(size_t)grow * N0 + gcol] = v;
    }
}

// ---------------------------------------------------------------------------
// Row-max pool: out[b, c0+c] = max_n in[b,n,c]
// ---------------------------------------------------------------------------
__global__ void k_rowmax(const float* __restrict__ in, float* __restrict__ out,
                         int C, int c0, int outC) {
    int t = blockIdx.x * blockDim.x + threadIdx.x;
    if (t >= BATCH * C) return;
    int b = t / C, c = t - b * C;
    const float* p = in + (size_t)b * NPTS * C + c;
    float m = -FLT_BIG;
    for (int n = 0; n < NPTS; ++n) m = fmaxf(m, p[(size_t)n * C]);
    out[b * outC + c0 + c] = m;
}

// ---------------------------------------------------------------------------
// T-Net final: inT = h256 @ Wo^T + bo + I(3); write to ws and d_out tail
// ---------------------------------------------------------------------------
__global__ void k_tnet_final(const float* __restrict__ h, const float* __restrict__ Wo,
                             const float* __restrict__ bo, float* __restrict__ inT,
                             float* __restrict__ out_tail) {
    int t = threadIdx.x;
    if (t >= BATCH * 9) return;
    int b = t / 9, e = t - b * 9;
    const float* a = h + b * 256;
    const float* w = Wo + e * 256;
    float s = bo[e];
    for (int k = 0; k < 256; ++k) s += a[k] * w[k];
    if ((e / 3) == (e % 3)) s += 1.f;
    inT[t] = s;
    out_tail[t] = s;
}

// x0[b,n,d] = sum_c x[b,n,c] * T[b,c,d]
__global__ void k_transform(const float* __restrict__ x, const float* __restrict__ T,
                            float* __restrict__ x0) {
    int t = blockIdx.x * blockDim.x + threadIdx.x;
    if (t >= BATCH * NPTS) return;
    int b = t >> 12;
    const float* xi = x + (size_t)t * 3;
    const float* Tb = T + b * 9;
    float o0 = xi[0] * Tb[0] + xi[1] * Tb[3] + xi[2] * Tb[6];
    float o1 = xi[0] * Tb[1] + xi[1] * Tb[4] + xi[2] * Tb[7];
    float o2 = xi[0] * Tb[2] + xi[1] * Tb[5] + xi[2] * Tb[8];
    float* o = x0 + (size_t)t * 3;
    o[0] = o0; o[1] = o1; o[2] = o2;
}

// ---------------------------------------------------------------------------
// kNN: one block (256 thr) per query point. Distances live in 16 registers per
// thread; top-32 via wave shfl argmin + 8-entry LDS combine (2 barriers/iter).
// ordering key: sum_c xj_c*(xj_c - 2*xi_c)  (== d minus per-row constant |xi|^2)
// ---------------------------------------------------------------------------
__global__ __launch_bounds__(256) void k_knn(const float* __restrict__ X, int C,
                                             int* __restrict__ idx) {
    __shared__ float xi[64];
    __shared__ float wval[8];
    __shared__ int   wind[8];
    __shared__ int   chosen;
    const int point = blockIdx.x;
    const int b = point >> 12, i = point & (NPTS - 1);
    const int tid = threadIdx.x;
    const float* Xb = X + (size_t)b * NPTS * C;
    for (int c = tid; c < C; c += 256) xi[c] = Xb[(size_t)i * C + c];
    __syncthreads();

    float dloc[16];
#pragma unroll
    for (int q = 0; q < 16; ++q) {
        int j = tid + (q << 8);
        const float* xj = Xb + (size_t)j * C;
        if (q < 15) __builtin_prefetch(Xb + (size_t)(j + 256) * C, 0, 1);
        float d = 0.f;
        for (int c = 0; c < C; ++c) { float v = xj[c]; d += v * (v - 2.f * xi[c]); }
        dloc[q] = d;
    }

    const int lane = tid & 31, wv = tid >> 5;
    for (int s = 0; s < KNN; ++s) {
        float best = FLT_BIG; int bq = 0;
#pragma unroll
        for (int q = 0; q < 16; ++q)
            if (dloc[q] < best) { best = dloc[q]; bq = q; }
        int bj = tid + (bq << 8);
#pragma unroll
        for (int off = 16; off > 0; off >>= 1) {
            float ov = __shfl_down(best, off, 32);
            int   oj = __shfl_down(bj, off, 32);
            if (ov < best) { best = ov; bj = oj; }
        }
        if (lane == 0) { wval[wv] = best; wind[wv] = bj; }
        __syncthreads();
        if (tid == 0) {
            float bb = wval[0]; int jj = wind[0];
#pragma unroll
            for (int w = 1; w < 8; ++w)
                if (wval[w] < bb) { bb = wval[w]; jj = wind[w]; }
            chosen = jj;
            idx[(size_t)point * KNN + s] = jj;
        }
        __syncthreads();
        int cj = chosen;
        bool mine = (cj & 255) == tid;
        int qq = cj >> 8;
#pragma unroll
        for (int q = 0; q < 16; ++q)
            if (mine && q == qq) dloc[q] = FLT_BIG;
    }
}

// ---------------------------------------------------------------------------
// EdgeConv1 (C=3, K=6): scalar, fused gather + conv + bn + lrelu + max over k
// ---------------------------------------------------------------------------
__global__ void k_ec1(const float* __restrict__ X, const int* __restrict__ idx,
                      const float* __restrict__ W, const float* __restrict__ g,
                      const float* __restrict__ be, float* __restrict__ out) {
    int t = blockIdx.x * blockDim.x + threadIdx.x;
    if (t >= BATCH * NPTS * 64) return;
    int o = t & 63, point = t >> 6;
    int b = point >> 12, i = point & (NPTS - 1);
    const float* Xb = X + (size_t)b * NPTS * 3;
    float xi0 = Xb[(size_t)i * 3], xi1 = Xb[(size_t)i * 3 + 1], xi2 = Xb[(size_t)i * 3 + 2];
    const float* w = W + o * 6;
    float w0 = w[0], w1 = w[1], w2 = w[2];
    float base = xi0 * w[3] + xi1 * w[4] + xi2 * w[5];
    float gg = g[o], bb = be[o];
    const int* id = idx + (size_t)point * KNN;
    float m = -FLT_BIG;
    for (int kk = 0; kk < KNN; ++kk) {
        int j = id[kk];
        const float* xj = Xb + (size_t)j * 3;
        float v = (xj[0] - xi0) * w0 + (xj[1] - xi1) * w1 + (xj[2] - xi2) * w2 + base;
        v = v * gg + bb;
        v = v > 0.f ? v : 0.01f * v;
        m = fmaxf(m, v);
    }
    out[(size_t)point * 64 + o] = m;
}

// ---------------------------------------------------------------------------
// EdgeConv (C=64 -> Cout): 4 points per block (one per wave, 128 thr).
// 32KB LDS A-tile (edge features as f16); B fragments read from pre-converted
// f16 weights in global (<=32KB, L0/L2 resident). Fused bn+lrelu+max-over-32.
// ---------------------------------------------------------------------------
__global__ __launch_bounds__(128) void k_edgeconv(
    const float* __restrict__ X, const int* __restrict__ idx,
    const _Float16* __restrict__ W16, const float* __restrict__ gamma,
    const float* __restrict__ beta, float* __restrict__ out, int Cout) {
    __shared__ _Float16 As[4 * 32 * 128];   // [pp][kk][c]  32KB
    __shared__ int nb[4 * KNN];
    const int pt0 = blockIdx.x * 4;
    const int tid = threadIdx.x;

    nb[tid] = idx[(size_t)pt0 * KNN + tid];   // 128 ints = 4 points x 32
    __syncthreads();

    for (int e = tid; e < 4 * 32 * 128; e += 128) {
        int pp = e >> 12, kk = (e >> 7) & 31, c = e & 127;
        int point = pt0 + pp;
        int b = point >> 12, i = point & (NPTS - 1);
        const float* Xb = X + (size_t)b * NPTS * 64;
        float v;
        if (c < 64) v = Xb[(size_t)nb[pp * 32 + kk] * 64 + c] - Xb[(size_t)i * 64 + c];
        else        v = Xb[(size_t)i * 64 + (c - 64)];
        As[e] = (_Float16)v;
    }
    __syncthreads();

    const int wave = tid >> 5, lane = tid & 31;
    const int half = lane >> 4, lr = lane & 15;
    const int point = pt0 + wave;
    const _Float16* Asp = As + wave * 32 * 128;
    const int ntiles = Cout >> 4;

    for (int nt = 0; nt < ntiles; ++nt) {
        v8f acc0 = {}, acc1 = {};   // kk rows 0-15 and 16-31
        for (int k0 = 0; k0 < 128; k0 += 32) {
            v16h a0, a1, bf;
#pragma unroll
            for (int j = 0; j < 8; ++j) {
                int kp = ((j < 4) ? 2 * j : 16 + 2 * (j - 4)) + half * 8;
                a0[2 * j]     = Asp[lr * 128 + k0 + kp];
                a0[2 * j + 1] = Asp[lr * 128 + k0 + kp + 1];
                a1[2 * j]     = Asp[(16 + lr) * 128 + k0 + kp];
                a1[2 * j + 1] = Asp[(16 + lr) * 128 + k0 + kp + 1];
                int kb = kp;
                const _Float16* wp = W16 + (size_t)(nt * 16 + lr) * 128 + k0 + kb;
                bf[2 * j]     = wp[0];
                bf[2 * j + 1] = wp[1];
            }
            acc0 = __builtin_amdgcn_wmma_f32_16x16x32_f16(false, a0, false, bf,
                                                          (short)0, acc0, false, false);
            acc1 = __builtin_amdgcn_wmma_f32_16x16x32_f16(false, a1, false, bf,
                                                          (short)0, acc1, false, false);
        }
        const float g  = gamma[nt * 16 + lr];
        const float bb = beta[nt * 16 + lr];
        float m = -FLT_BIG;
#pragma unroll
        for (int ii = 0; ii < 8; ++ii) {
            float v0 = acc0[ii] * g + bb; v0 = v0 > 0.f ? v0 : 0.01f * v0;
            float v1 = acc1[ii] * g + bb; v1 = v1 > 0.f ? v1 : 0.01f * v1;
            m = fmaxf(m, fmaxf(v0, v1));
        }
        float o = __shfl_xor(m, 16, 32);   // lane L and L+16 hold the same column
        m = fmaxf(m, o);
        if (lane < 16) out[(size_t)point * Cout + nt * 16 + lane] = m;
    }
}

// ---------------------------------------------------------------------------
extern "C" void kernel_launch(void* const* d_in, const int* in_sizes, int n_in,
                              void* d_out, int out_size, void* d_ws, size_t ws_size,
                              hipStream_t stream) {
    (void)in_sizes; (void)n_in; (void)out_size; (void)ws_size;
    // --- inputs (setup_inputs insertion order) ---
    const float* x    = (const float*)d_in[0];
    const float* W1   = (const float*)d_in[1];
    const float* g1   = (const float*)d_in[2];
    const float* b1   = (const float*)d_in[3];
    const float* W2   = (const float*)d_in[4];
    const float* g2   = (const float*)d_in[5];
    const float* b2   = (const float*)d_in[6];
    const float* W3c  = (const float*)d_in[7];
    const float* g3   = (const float*)d_in[8];
    const float* b3   = (const float*)d_in[9];
    const float* Wf1  = (const float*)d_in[10];
    const float* bf1  = (const float*)d_in[11];
    const float* gf1  = (const float*)d_in[12];
    const float* bef1 = (const float*)d_in[13];
    const float* Wf2  = (const float*)d_in[14];
    const float* bf2  = (const float*)d_in[15];
    const float* gf2  = (const float*)d_in[16];
    const float* bef2 = (const float*)d_in[17];
    const float* Wo   = (const float*)d_in[18];
    const float* bo   = (const float*)d_in[19];
    const float* We1  = (const float*)d_in[20];
    const float* ge1  = (const float*)d_in[21];
    const float* be1  = (const float*)d_in[22];
    const float* We2  = (const float*)d_in[23];
    const float* ge2  = (const float*)d_in[24];
    const float* be2  = (const float*)d_in[25];
    const float* We3  = (const float*)d_in[26];
    const float* ge3  = (const float*)d_in[27];
    const float* be3  = (const float*)d_in[28];
    const float* We4  = (const float*)d_in[29];
    const float* ge4  = (const float*)d_in[30];
    const float* be4  = (const float*)d_in[31];
    const float* Wh1  = (const float*)d_in[32];
    const float* bh1  = (const float*)d_in[33];
    const float* Wh2  = (const float*)d_in[34];
    const float* bh2  = (const float*)d_in[35];
    const float* Wh3  = (const float*)d_in[36];
    const float* bh3  = (const float*)d_in[37];
    const float* Wh4  = (const float*)d_in[38];
    const float* bh4  = (const float*)d_in[39];

    float* outp = (float*)d_out;      // [0,160): logits [4,40]; [160,196): inT [4,3,3]
    float* ws = (float*)d_ws;
    const int M = BATCH * NPTS;       // 16384

    // small persistent buffers
    float* g1024   = ws + 0;
    float* h512    = ws + 4096;
    float* h256    = ws + 6144;
    float* inT     = ws + 7168;
    float* g320    = ws + 7232;
    float* gh1024  = ws + 8512;
    float* gh512   = ws + 12608;
    float* gh128   = ws + 14656;
    _Float16* We2h = (_Float16*)(ws + 15168);   // 64*128  halfs
    _Float16* We3h = (_Float16*)(ws + 19264);   // 64*128  halfs
    _Float16* We4h = (_Float16*)(ws + 23360);   // 128*128 halfs
    float* big     = ws + 32768;
    // phase A (T-Net), aliased by phase B (EdgeConv)
    float* h1 = big;
    float* h2 = h1 + (size_t)M * 64;
    float* h3 = h2 + (size_t)M * 128;
    float* x0 = big;
    float* x1 = x0 + (size_t)M * 3;
    float* x2 = x1 + (size_t)M * 64;
    float* x3 = x2 + (size_t)M * 64;
    float* x4 = x3 + (size_t)M * 64;
    int*  idx = (int*)(x4 + (size_t)M * 128);

    // ---- weight pre-conversion (f32 -> f16, once) ----
    k_cvt_f16<<<(64 * 128 + 255) / 256, 256, 0, stream>>>(We2, We2h, 64 * 128);
    k_cvt_f16<<<(64 * 128 + 255) / 256, 256, 0, stream>>>(We3, We3h, 64 * 128);
    k_cvt_f16<<<(128 * 128 + 255) / 256, 256, 0, stream>>>(We4, We4h, 128 * 128);

    // ---- T-Net ----
    k_linear<<<(M * 64 + 255) / 256, 256, 0, stream>>>(x, W1, nullptr, g1, b1, h1,
                                                       M, 64, 3, ACT_RELU);
    k_gemm_wmma<<<dim3(M / 64, 128 / 16), 128, 0, stream>>>(h1, W2, g2, b2, h2,
                                                            M, 128, 64, ACT_RELU);
    k_gemm_wmma<<<dim3(M / 64, 1024 / 16), 128, 0, stream>>>(h2, W3c, g3, b3, h3,
                                                             M, 1024, 128, ACT_RELU);
    k_rowmax<<<(BATCH * 1024 + 255) / 256, 256, 0, stream>>>(h3, g1024, 1024, 0, 1024);
    k_linear<<<(BATCH * 512 + 255) / 256, 256, 0, stream>>>(g1024, Wf1, bf1, gf1, bef1,
                                                            h512, BATCH, 512, 1024, ACT_RELU);
    k_linear<<<(BATCH * 256 + 255) / 256, 256, 0, stream>>>(h512, Wf2, bf2, gf2, bef2,
                                                            h256, BATCH, 256, 512, ACT_RELU);
    k_tnet_final<<<1, 64, 0, stream>>>(h256, Wo, bo, inT, outp + 160);
    k_transform<<<(M + 255) / 256, 256, 0, stream>>>(x, inT, x0);

    // ---- EdgeConv stack ----
    k_knn<<<M, 256, 0, stream>>>(x0, 3, idx);
    k_ec1<<<(M * 64 + 255) / 256, 256, 0, stream>>>(x0, idx, We1, ge1, be1, x1);

    k_knn<<<M, 256, 0, stream>>>(x1, 64, idx);
    k_edgeconv<<<M / 4, 128, 0, stream>>>(x1, idx, We2h, ge2, be2, x2, 64);

    k_knn<<<M, 256, 0, stream>>>(x2, 64, idx);
    k_edgeconv<<<M / 4, 128, 0, stream>>>(x2, idx, We3h, ge3, be3, x3, 64);

    k_knn<<<M, 256, 0, stream>>>(x3, 64, idx);
    k_edgeconv<<<M / 4, 128, 0, stream>>>(x3, idx, We4h, ge4, be4, x4, 128);

    // ---- Global pool (concat x1..x4 -> 320) ----
    k_rowmax<<<1, 256, 0, stream>>>(x1, g320, 64, 0, 320);
    k_rowmax<<<1, 256, 0, stream>>>(x2, g320, 64, 64, 320);
    k_rowmax<<<1, 256, 0, stream>>>(x3, g320, 64, 128, 320);
    k_rowmax<<<2, 256, 0, stream>>>(x4, g320, 128, 192, 320);

    // ---- Head ----
    k_linear<<<(BATCH * 1024 + 255) / 256, 256, 0, stream>>>(g320, Wh1, bh1, nullptr, nullptr,
                                                             gh1024, BATCH, 1024, 320, ACT_LRELU);
    k_linear<<<(BATCH * 512 + 255) / 256, 256, 0, stream>>>(gh1024, Wh2, bh2, nullptr, nullptr,
                                                            gh512, BATCH, 512, 1024, ACT_LRELU);
    k_linear<<<(BATCH * 128 + 255) / 256, 256, 0, stream>>>(gh512, Wh3, bh3, nullptr, nullptr,
                                                            gh128, BATCH, 128, 512, ACT_LRELU);
    k_linear<<<1, 256, 0, stream>>>(gh128, Wh4, bh4, nullptr, nullptr,
                                    outp, BATCH, 40, 128, ACT_NONE);
}
